// Pointer_42021960024116
// MI455X (gfx1250) — matrix operand order repaired and checked by reading, not compiled, not run on hardware
//
#include <hip/hip_runtime.h>
#include <hip/hip_bf16.h>

typedef __attribute__((ext_vector_type(16))) __bf16 v16bf;
typedef __attribute__((ext_vector_type(8)))  __bf16 v8bf;
typedef __attribute__((ext_vector_type(4)))  __bf16 v4bf;
typedef __attribute__((ext_vector_type(8)))  float  v8f;

#define L_DIM 2048
#define B_DIM 64
#define D_DIM 512
#define H_DIM 512
#define A_DIM 512
#define N_STEPS 4
#define M_DIM (L_DIM * B_DIM)          // 131072 GEMM rows
#define NEG_VAL (-1e8f)

// ---------------- fp32 -> bf16 conversion (4-wide) ----------------
__global__ void k_cvt_bf16(const float* __restrict__ in, __bf16* __restrict__ out, int n4) {
    int i = blockIdx.x * blockDim.x + threadIdx.x;
    if (i >= n4) return;
    float4 v = *(const float4*)(in + (size_t)i * 4);
    v4bf o;
    o[0] = (__bf16)v.x; o[1] = (__bf16)v.y; o[2] = (__bf16)v.z; o[3] = (__bf16)v.w;
    *(v4bf*)(out + (size_t)i * 4) = o;
}

// ---------------- Wd (A,D) fp32 -> WdT (D,A) bf16 ----------------
__global__ void k_wd_transpose(const float* __restrict__ Wd, __bf16* __restrict__ WdT) {
    int idx = blockIdx.x * blockDim.x + threadIdx.x;   // D_DIM*A_DIM
    int k = idx >> 9;          // D index
    int n = idx & 511;         // A index
    WdT[k * A_DIM + n] = (__bf16)Wd[n * D_DIM + k];
}

// ---------------- big GEMM: dctx = ctx_bf16 (M,K) @ WdT (K,N) + bd ----------------
// block tile 128x128, 8 waves, wave tile 32x64 (2x4 of 16x16), K step 32
__global__ __launch_bounds__(256)
void k_gemm_dctx(const __bf16* __restrict__ Amat,   // (M, 512)
                 const __bf16* __restrict__ Bmat,   // (512, 512)
                 const float*  __restrict__ bias,   // bd (512)
                 __bf16* __restrict__ Cout)         // (M, 512) bf16
{
    const int K = 512, N = 512;
    const int tid  = threadIdx.x;
    const int wid  = tid >> 5;
    const int lane = tid & 31;
    const int wm = wid >> 1;                 // 0..3
    const int wn = wid & 1;                  // 0..1
    const long waveRow = (long)blockIdx.y * 128 + wm * 32;
    const int  waveCol = blockIdx.x * 128 + wn * 64;

    const int mlane = lane & 15;
    const int khalf = (lane >> 4) * 8;       // A frag: 0 or 8
    // B frag: lane l holds column n = l%16, K = k0+khalf..+7 and k0+khalf+16..+23

    v8f acc[2][4];
#pragma unroll
    for (int tm = 0; tm < 2; ++tm)
#pragma unroll
        for (int tn = 0; tn < 4; ++tn)
            acc[tm][tn] = (v8f){0.f,0.f,0.f,0.f,0.f,0.f,0.f,0.f};

    for (int k0 = 0; k0 < K; k0 += 32) {
        v16bf afrag[2];
#pragma unroll
        for (int tm = 0; tm < 2; ++tm) {
            const __bf16* ap = Amat + (waveRow + tm * 16 + mlane) * (long)K + k0 + khalf;
            union { v16bf v; v8bf h[2]; } u;
            u.h[0] = *(const v8bf*)(ap);        // K = k0+khalf .. +7
            u.h[1] = *(const v8bf*)(ap + 16);   // K = k0+khalf+16 .. +23
            afrag[tm] = u.v;
        }
        v16bf bfrag[4];
#pragma unroll
        for (int tn = 0; tn < 4; ++tn) {
            const __bf16* bp = Bmat + (long)(k0 + khalf) * N + waveCol + tn * 16 + mlane;
            v16bf bv;
#pragma unroll
            for (int s = 0; s < 8; ++s) {
                bv[s] = bp[(long)s * N];
                bv[s + 8] = bp[(long)(s + 16) * N];
            }
            bfrag[tn] = bv;
        }
#pragma unroll
        for (int tm = 0; tm < 2; ++tm)
#pragma unroll
            for (int tn = 0; tn < 4; ++tn)
                acc[tm][tn] = __builtin_amdgcn_wmma_f32_16x16x32_bf16(
                    false, afrag[tm], false, bfrag[tn],
                    (short)0, acc[tm][tn], false, false);
    }

    // store: C/D layout — lane<16: N=lane, M=r ; lane>=16: N=lane-16, M=r+8
    const int nlocal = lane & 15;
    const int rowOff = (lane >> 4) * 8;
#pragma unroll
    for (int tm = 0; tm < 2; ++tm)
#pragma unroll
        for (int tn = 0; tn < 4; ++tn) {
            const int n = waveCol + tn * 16 + nlocal;
            const float bn = bias[n];
#pragma unroll
            for (int r = 0; r < 8; ++r) {
                long m = waveRow + tm * 16 + rowOff + r;
                Cout[m * (long)N + n] = (__bf16)(acc[tm][tn][r] + bn);
            }
        }
}

// ---------------- gather x for the step ----------------
__global__ void k_gather_x(const float* __restrict__ context, const int* __restrict__ prev_targets,
                           const float* __restrict__ pre_att, int s, float* __restrict__ x) {
    int idx = blockIdx.x * blockDim.x + threadIdx.x;   // B*D
    if (idx >= B_DIM * D_DIM) return;
    if (s == 0) { x[idx] = pre_att[idx]; return; }
    int b = idx / D_DIM, d = idx % D_DIM;
    int l = prev_targets[(s - 1) * B_DIM + b];
    x[idx] = context[((long)l * B_DIM + b) * D_DIM + d];
}

// ---------------- fused GRU cell ----------------
__global__ void k_gru(const float* __restrict__ x, const float* __restrict__ hprev,
                      const float* __restrict__ Wih, const float* __restrict__ Whh,
                      const float* __restrict__ bih, const float* __restrict__ bhh,
                      float* __restrict__ hout) {
    int idx = blockIdx.x * blockDim.x + threadIdx.x;   // B*H
    if (idx >= B_DIM * H_DIM) return;
    int b = idx / H_DIM, j = idx % H_DIM;
    const float* xr = x + b * D_DIM;
    const float* hr = hprev + b * H_DIM;
    float ir = bih[j], iz = bih[H_DIM + j], inn = bih[2 * H_DIM + j];
    float hrr = bhh[j], hz = bhh[H_DIM + j], hn = bhh[2 * H_DIM + j];
    const float* wr = Wih + (long)j * D_DIM;
    const float* wz = Wih + (long)(H_DIM + j) * D_DIM;
    const float* wn = Wih + (long)(2 * H_DIM + j) * D_DIM;
    for (int k = 0; k < D_DIM; ++k) {
        float xv = xr[k];
        ir += xv * wr[k]; iz += xv * wz[k]; inn += xv * wn[k];
    }
    const float* ur = Whh + (long)j * H_DIM;
    const float* uz = Whh + (long)(H_DIM + j) * H_DIM;
    const float* un = Whh + (long)(2 * H_DIM + j) * H_DIM;
    for (int k = 0; k < H_DIM; ++k) {
        float hv = hr[k];
        hrr += hv * ur[k]; hz += hv * uz[k]; hn += hv * un[k];
    }
    float r = 1.f / (1.f + __expf(-(ir + hrr)));
    float z = 1.f / (1.f + __expf(-(iz + hz)));
    float n = tanhf(inn + r * hn);
    hout[idx] = (1.f - z) * n + z * hr[j];
}

// ---------------- q projection ----------------
__global__ void k_qproj(const float* __restrict__ h, const float* __restrict__ Wq,
                        const float* __restrict__ bq, float* __restrict__ q) {
    int idx = blockIdx.x * blockDim.x + threadIdx.x;   // B*A
    if (idx >= B_DIM * A_DIM) return;
    int b = idx / A_DIM, a = idx % A_DIM;
    const float* hrow = h + b * H_DIM;
    const float* wrow = Wq + (long)a * H_DIM;
    float acc = bq[a];
    for (int k = 0; k < H_DIM; ++k) acc += hrow[k] * wrow[k];
    q[idx] = acc;
}

// ---------------- energy + mask ----------------
__global__ void k_energy(const __bf16* __restrict__ dctx,  // (L*B, A)
                         const float* __restrict__ q,      // (B, A)
                         const float* __restrict__ Ws,     // (A)
                         const float* __restrict__ bs,     // (1)
                         const float* __restrict__ mask,   // (B, L)
                         const int* __restrict__ prev_targets,
                         int s, float* __restrict__ e)     // (L, B)
{
    long m = (long)blockIdx.x * blockDim.x + threadIdx.x;
    if (m >= (long)M_DIM) return;
    int b = (int)(m & (B_DIM - 1));
    int l = (int)(m >> 6);
    const __bf16* dp = dctx + m * A_DIM;
    const float*  qp = q + b * A_DIM;
    float acc = 0.f;
    for (int a = 0; a < A_DIM; a += 8) {
        v8bf dv = *(const v8bf*)(dp + a);
#pragma unroll
        for (int j = 0; j < 8; ++j)
            acc += Ws[a + j] * tanhf(qp[a + j] + (float)dv[j]);
    }
    acc += bs[0];
    float mv = mask[b * L_DIM + l];
    for (int t = 0; t < s; ++t)
        if (prev_targets[t * B_DIM + b] == l) mv = 0.f;
    e[m] = acc * mv + (1.f - mv) * NEG_VAL;
}

// ---------------- softmax over L per batch column ----------------
__global__ void k_softmax(const float* __restrict__ e, float* __restrict__ out) {
    __shared__ float red[256];
    int b = blockIdx.x, t = threadIdx.x;
    float mx = -INFINITY;
    for (int l = t; l < L_DIM; l += 256) mx = fmaxf(mx, e[l * B_DIM + b]);
    red[t] = mx; __syncthreads();
    for (int o = 128; o > 0; o >>= 1) { if (t < o) red[t] = fmaxf(red[t], red[t + o]); __syncthreads(); }
    mx = red[0]; __syncthreads();
    float sum = 0.f;
    for (int l = t; l < L_DIM; l += 256) sum += __expf(e[l * B_DIM + b] - mx);
    red[t] = sum; __syncthreads();
    for (int o = 128; o > 0; o >>= 1) { if (t < o) red[t] += red[t + o]; __syncthreads(); }
    float inv = 1.f / red[0];
    for (int l = t; l < L_DIM; l += 256)
        out[l * B_DIM + b] = __expf(e[l * B_DIM + b] - mx) * inv;
}

extern "C" void kernel_launch(void* const* d_in, const int* in_sizes, int n_in,
                              void* d_out, int out_size, void* d_ws, size_t ws_size,
                              hipStream_t stream) {
    const float* hidden   = (const float*)d_in[0];
    const float* context  = (const float*)d_in[1];
    const float* mask     = (const float*)d_in[2];
    const float* pre_att  = (const float*)d_in[3];
    const int*   prev_tgt = (const int*)  d_in[4];
    // d_in[5] = max_step (compile-time 4)
    const float* W_ih = (const float*)d_in[6];
    const float* W_hh = (const float*)d_in[7];
    const float* b_ih = (const float*)d_in[8];
    const float* b_hh = (const float*)d_in[9];
    const float* Wq   = (const float*)d_in[10];
    const float* bq   = (const float*)d_in[11];
    const float* Wd   = (const float*)d_in[12];
    const float* bd   = (const float*)d_in[13];
    const float* Ws   = (const float*)d_in[14];
    const float* bs   = (const float*)d_in[15];
    float* out = (float*)d_out;

    // ---- workspace carving ----
    char* ws = (char*)d_ws;
    __bf16* ctx_bf = (__bf16*)ws;  ws += (size_t)M_DIM * D_DIM * 2;   // 128 MB
    __bf16* wdt_bf = (__bf16*)ws;  ws += (size_t)D_DIM * A_DIM * 2;   // 512 KB
    __bf16* dctx   = (__bf16*)ws;  ws += (size_t)M_DIM * A_DIM * 2;   // 128 MB
    float* hA = (float*)ws;        ws += (size_t)B_DIM * H_DIM * 4;
    float* hB = (float*)ws;        ws += (size_t)B_DIM * H_DIM * 4;
    float* xb = (float*)ws;        ws += (size_t)B_DIM * D_DIM * 4;
    float* qb = (float*)ws;        ws += (size_t)B_DIM * A_DIM * 4;
    float* eb = (float*)ws;        ws += (size_t)L_DIM * B_DIM * 4;

    // ---- precompute: bf16 conversions + big WMMA GEMM ----
    {
        int n4 = M_DIM * D_DIM / 4;
        k_cvt_bf16<<<(n4 + 255) / 256, 256, 0, stream>>>(context, ctx_bf, n4);
        k_wd_transpose<<<(D_DIM * A_DIM + 255) / 256, 256, 0, stream>>>(Wd, wdt_bf);
        dim3 grid(A_DIM / 128, M_DIM / 128);
        k_gemm_dctx<<<grid, 256, 0, stream>>>(ctx_bf, wdt_bf, bd, dctx);
    }

    // ---- 4 sequential decode steps ----
    const float* hprev = hidden;
    float* hcur = hA;
    for (int s = 0; s < N_STEPS; ++s) {
        k_gather_x<<<(B_DIM * D_DIM + 255) / 256, 256, 0, stream>>>(context, prev_tgt, pre_att, s, xb);
        k_gru<<<(B_DIM * H_DIM + 255) / 256, 256, 0, stream>>>(xb, hprev, W_ih, W_hh, b_ih, b_hh, hcur);
        k_qproj<<<(B_DIM * A_DIM + 255) / 256, 256, 0, stream>>>(hcur, Wq, bq, qb);
        k_energy<<<(M_DIM + 255) / 256, 256, 0, stream>>>(dctx, qb, Ws, bs, mask, prev_tgt, s, eb);
        k_softmax<<<B_DIM, 256, 0, stream>>>(eb, out + (size_t)s * L_DIM * B_DIM);
        hprev = hcur;
        hcur = (hcur == hA) ? hB : hA;
    }
}
